// RNNFeaturizer_56392920596778
// MI455X (gfx1250) — matrix operand order repaired and checked by reading, not compiled, not run on hardware
//
#include <hip/hip_runtime.h>
#include <hip/hip_bf16.h>

// Problem constants (from reference): T=1024, B=64, V=256, E=256, H=1024
#define T_STEPS 1024
#define BATCH   64
#define EDIM    256
#define HDIM    1024
#define G4H     (4 * HDIM)

typedef __attribute__((ext_vector_type(16))) __bf16 v16bf;
typedef __attribute__((ext_vector_type(8)))  __bf16 v8bf;
typedef __attribute__((ext_vector_type(8)))  float  v8f;

__device__ __forceinline__ unsigned short f2bf(float f) {
    union { float f; unsigned u; } v; v.f = f;
    unsigned r = v.u + 0x7FFFu + ((v.u >> 16) & 1u);   // round-to-nearest-even
    return (unsigned short)(r >> 16);
}

__device__ __forceinline__ float fsig(float x) {
    return 1.0f / (1.0f + __expf(-x));
}
__device__ __forceinline__ float ftanh(float x) {
    float e = __expf(-2.0f * x);
    return (1.0f - e) / (1.0f + e);
}

// ---------------------------------------------------------------------------
// Prep: fp32 -> bf16 weight conversion, bias fusion, zero h0 / c0.
// Grid covers G4H*HDIM elements (largest array = W_hh).
// ---------------------------------------------------------------------------
__global__ void lstm_prep_kernel(const float* __restrict__ Wih,
                                 const float* __restrict__ Whh,
                                 const float* __restrict__ bih,
                                 const float* __restrict__ bhh,
                                 unsigned short* __restrict__ Wihb,
                                 unsigned short* __restrict__ Whhb,
                                 float* __restrict__ bias,
                                 unsigned short* __restrict__ h0,
                                 float* __restrict__ c0) {
    int idx = blockIdx.x * blockDim.x + threadIdx.x;
    if (idx < G4H * HDIM) Whhb[idx] = f2bf(Whh[idx]);
    if (idx < G4H * EDIM) Wihb[idx] = f2bf(Wih[idx]);
    if (idx < G4H)        bias[idx] = bih[idx] + bhh[idx];
    if (idx < BATCH * HDIM) { h0[idx] = 0; c0[idx] = 0.0f; }
}

// ---------------------------------------------------------------------------
// Embed: xb[t,b,e] = bf16(emb[tokens[t,b], e])   (one-time gather, 32 MB)
// ---------------------------------------------------------------------------
__global__ void lstm_embed_kernel(const int* __restrict__ tokens,
                                  const float* __restrict__ emb,
                                  unsigned short* __restrict__ xb) {
    int idx = blockIdx.x * blockDim.x + threadIdx.x;   // over T*B*E = 16.7M
    int e  = idx & (EDIM - 1);
    int tb = idx >> 8;                                 // EDIM == 256
    int tok = tokens[tb];
    xb[idx] = f2bf(emb[tok * EDIM + e]);
}

// ---------------------------------------------------------------------------
// GEMM accumulate: acc[2][4] += A_tile x B_tile over K, bf16 WMMA 16x16x32.
//   A: row-major [64, ldka] bf16 (activations)
//   B: "column n of B" = row (gate_row_base + n) of W, row-major [4H, ldkb]
// Fragment lane layouts used here (contiguous k, same order for A and B):
//   A 16x32: lanes 0-15 row M=l hold K k0..k0+15;
//            lanes 16-31 row M=l-16 hold K k0+16..k0+31
//   B 32x16: lanes 0-15 col N=l hold K k0..k0+15; lanes 16-31 K k0+16..k0+31
// ---------------------------------------------------------------------------
__device__ __forceinline__ void gemm_acc(const unsigned short* __restrict__ Abase,
                                         int ldka,
                                         const unsigned short* __restrict__ Bbase,
                                         int ldkb, int K,
                                         int mpair, int lmod, bool hi,
                                         int gate_row_base,
                                         v8f acc[2][4]) {
    const __bf16* A  = reinterpret_cast<const __bf16*>(Abase);
    const __bf16* Bp = reinterpret_cast<const __bf16*>(Bbase);

    const long arow0 = (long)(mpair * 32 +  0 + lmod) * ldka;
    const long arow1 = (long)(mpair * 32 + 16 + lmod) * ldka;
    const int  akoff = hi ? 16 : 0;
    const int  bkoff = hi ? 16 : 0;

    long brow[4];
#pragma unroll
    for (int n = 0; n < 4; ++n)
        brow[n] = (long)(gate_row_base + n * 16 + lmod) * ldkb;

    for (int k0 = 0; k0 < K; k0 += 32) {
        // A fragments (two 16-byte chunks per lane, see layout above)
        v8bf a0lo = *reinterpret_cast<const v8bf*>(A + arow0 + k0 + akoff);
        v8bf a0hi = *reinterpret_cast<const v8bf*>(A + arow0 + k0 + akoff + 8);
        v8bf a1lo = *reinterpret_cast<const v8bf*>(A + arow1 + k0 + akoff);
        v8bf a1hi = *reinterpret_cast<const v8bf*>(A + arow1 + k0 + akoff + 8);
        v16bf a0 = __builtin_shufflevector(a0lo, a0hi, 0,1,2,3,4,5,6,7,8,9,10,11,12,13,14,15);
        v16bf a1 = __builtin_shufflevector(a1lo, a1hi, 0,1,2,3,4,5,6,7,8,9,10,11,12,13,14,15);

        if (k0 + 32 < K) {  // L2-resident weights: keep the stream warm
            __builtin_prefetch(Bp + brow[0] + k0 + 32 + bkoff, 0, 0);
        }

#pragma unroll
        for (int n = 0; n < 4; ++n) {
            v16bf b = *reinterpret_cast<const v16bf*>(Bp + brow[n] + k0 + bkoff);
            acc[0][n] = __builtin_amdgcn_wmma_f32_16x16x32_bf16(
                false, a0, false, b, (short)0, acc[0][n], false, false);
            acc[1][n] = __builtin_amdgcn_wmma_f32_16x16x32_bf16(
                false, a1, false, b, (short)0, acc[1][n], false, false);
        }
    }
}

// ---------------------------------------------------------------------------
// One LSTM timestep. 16 blocks x 256 threads (8 waves).
// Block b owns H-columns [64*b, 64*b+64): computes i/f/g/o gate tiles (64x64
// each) with WMMA, then the fused elementwise cell update.
// Wave w: gate = w&3, m-pair = w>>2 (rows {0..31} or {32..63} of batch).
// ---------------------------------------------------------------------------
__launch_bounds__(256, 2)
__global__ void lstm_step_kernel(const unsigned short* __restrict__ xb_t,
                                 const unsigned short* __restrict__ h_in,
                                 unsigned short* __restrict__ h_out,
                                 const unsigned short* __restrict__ Wihb,
                                 const unsigned short* __restrict__ Whhb,
                                 const float* __restrict__ bias,
                                 float* __restrict__ c_state,
                                 const int* __restrict__ seq_len,
                                 float* __restrict__ out,
                                 int t) {
    __shared__ float lds_g[4][64][64];   // 64 KB: pre-activation gate tiles

    const int tid   = threadIdx.x;
    const int wave  = tid >> 5;
    const int lane  = tid & 31;
    const int gate  = wave & 3;
    const int mpair = wave >> 2;
    const int lmod  = lane & 15;
    const bool hi   = lane >= 16;
    const int jblock = blockIdx.x * 64;
    const int grow   = gate * HDIM + jblock;   // first W row of this gate tile

    v8f acc[2][4];
#pragma unroll
    for (int s = 0; s < 2; ++s)
#pragma unroll
        for (int n = 0; n < 4; ++n)
            acc[s][n] = (v8f){0.f, 0.f, 0.f, 0.f, 0.f, 0.f, 0.f, 0.f};

    // gates += x_t @ W_ih^T   (K = 256)
    gemm_acc(xb_t, EDIM, Wihb, EDIM, EDIM, mpair, lmod, hi, grow, acc);
    // gates += h   @ W_hh^T   (K = 1024)
    gemm_acc(h_in, HDIM, Whhb, HDIM, HDIM, mpair, lmod, hi, grow, acc);

    // Spill gate tiles (+bias) to LDS. C layout (7.12.2): VGPR r, lane l ->
    // (m = r + (l>=16 ? 8 : 0), n = l%16) within a 16x16 subtile.
#pragma unroll
    for (int n = 0; n < 4; ++n) {
        float bv = bias[grow + n * 16 + lmod];
#pragma unroll
        for (int s = 0; s < 2; ++s) {
            int mbase = mpair * 32 + s * 16 + (hi ? 8 : 0);
#pragma unroll
            for (int r = 0; r < 8; ++r)
                lds_g[gate][mbase + r][n * 16 + lmod] = acc[s][n][r] + bv;
        }
    }
    __syncthreads();

    // Fused LSTM cell update: each thread handles 16 (b, j) elements.
    for (int e = tid; e < 64 * 64; e += 256) {
        int b  = e >> 6;
        int jl = e & 63;
        int j  = jblock + jl;
        float iv = fsig(lds_g[0][b][jl]);
        float fv = fsig(lds_g[1][b][jl]);
        float gv = ftanh(lds_g[2][b][jl]);
        float ov = fsig(lds_g[3][b][jl]);
        float co = c_state[b * HDIM + j];
        float cn = fv * co + iv * gv;
        float hn = ov * ftanh(cn);
        c_state[b * HDIM + j] = cn;
        h_out[b * HDIM + j]   = f2bf(hn);
        bool valid = (t == 0) || (t < seq_len[b]);
        if (valid) out[b * HDIM + j] = cn;   // last_c written in place
    }
}

// ---------------------------------------------------------------------------
extern "C" void kernel_launch(void* const* d_in, const int* in_sizes, int n_in,
                              void* d_out, int out_size, void* d_ws, size_t ws_size,
                              hipStream_t stream) {
    (void)in_sizes; (void)n_in; (void)out_size; (void)ws_size;

    const int*   tokens  = (const int*)  d_in[0];   // [T, B]
    const int*   seq_len = (const int*)  d_in[1];   // [B]
    const float* emb     = (const float*)d_in[2];   // [V, E]
    const float* Wih     = (const float*)d_in[3];   // [4H, E]
    const float* Whh     = (const float*)d_in[4];   // [4H, H]
    const float* bih     = (const float*)d_in[5];   // [4H]
    const float* bhh     = (const float*)d_in[6];   // [4H]
    float*       out     = (float*)d_out;           // [B, H] = last_c

    // Workspace carve-up (256B aligned).  Total ~44.6 MB.
    char* ws = (char*)d_ws;
    size_t off = 0;
    auto carve = [&](size_t bytes) {
        void* p = ws + off;
        off = (off + bytes + 255) & ~(size_t)255;
        return p;
    };
    unsigned short* Whhb  = (unsigned short*)carve((size_t)G4H * HDIM * 2);   // 8 MB
    unsigned short* Wihb  = (unsigned short*)carve((size_t)G4H * EDIM * 2);   // 2 MB
    float*          biasf = (float*)         carve((size_t)G4H * 4);          // 16 KB
    unsigned short* xb    = (unsigned short*)carve((size_t)T_STEPS * BATCH * EDIM * 2); // 32 MB
    unsigned short* hA    = (unsigned short*)carve((size_t)BATCH * HDIM * 2); // 128 KB
    unsigned short* hB    = (unsigned short*)carve((size_t)BATCH * HDIM * 2); // 128 KB
    float*          cbuf  = (float*)         carve((size_t)BATCH * HDIM * 4); // 256 KB

    // 1) Weights -> bf16, bias fusion, zero h0/c0
    {
        int n = G4H * HDIM;
        lstm_prep_kernel<<<(n + 255) / 256, 256, 0, stream>>>(
            Wih, Whh, bih, bhh, Wihb, Whhb, biasf, hA, cbuf);
    }
    // 2) Embedding gather -> bf16
    {
        int n = T_STEPS * BATCH * EDIM;
        lstm_embed_kernel<<<n / 256, 256, 0, stream>>>(tokens, emb, xb);
    }
    // 3) Recurrence: 1024 graph-captured step launches, h double-buffered
    for (int t = 0; t < T_STEPS; ++t) {
        const unsigned short* hin  = (t & 1) ? hB : hA;
        unsigned short*       hout = (t & 1) ? hA : hB;
        lstm_step_kernel<<<16, 256, 0, stream>>>(
            xb + (size_t)t * BATCH * EDIM, hin, hout,
            Wihb, Whhb, biasf, cbuf, seq_len, out, t);
    }
}
